// SlowFastEC_85478439125031
// MI455X (gfx1250) — matrix operand order repaired and checked by reading, not compiled, not run on hardware
//
#include <hip/hip_runtime.h>
#include <hip/hip_bf16.h>

// ---------------------------------------------------------------------------
// SlowFastEC on MI455X (gfx1250): all GEMMs via v_wmma_f32_16x16x32_bf16.
//   k_gi  : gi[t][b][192] = slow_frames @ W_ih^T + b_ih   (parallel, full chip)
//   k_gru : sequential GRU; 8 WGs x 4 waves; W_hh pinned in VGPRs;
//           ping-pong LDS h (1 barrier/step); gi software-pipelined
//   k_cs  : cs[t][b][32] = hs @ W_cs^T + b_cs              (parallel)
//   k_mlp : y = relu([fast||cond] @ W1^T + b1) @ W2^T + b2 (parallel)
//   k_ola : deterministic 2-term overlap-add gather (no atomics)
// ---------------------------------------------------------------------------

typedef __attribute__((ext_vector_type(16))) __bf16 v16bf;
typedef __attribute__((ext_vector_type(8)))  float  v8f;

#define LF     32
#define HOPC   16
#define DELTAC 3
#define LS     96
#define CONDC  32
#define GHC    64
#define BBATCH 128
#define TLEN   48000
#define NF     2999          // (T - LF)/HOP + 1
#define NS     999           // (T - 96)/48 + 1
#define HOP_S  48
#define NG     192           // 3*GH

static __device__ __forceinline__ v8f wmma_bf16(v16bf a, v16bf b, v8f c) {
  return __builtin_amdgcn_wmma_f32_16x16x32_bf16(false, a, false, b,
                                                 (short)0, c, false, false);
}

static __device__ __forceinline__ float sigmoidf_(float x) {
  return 1.0f / (1.0f + __expf(-x));
}

// A fragment (16x32 bf16, M x K): lane l holds row m = l&15.
// lanes 0-15:  elems 0..7 -> K=0..7,   elems 8..15 -> K=16..23
// lanes 16-31: elems 0..7 -> K=8..15,  elems 8..15 -> K=24..31
// rowp points at K=0 of row m (f32 source).
static __device__ __forceinline__ v16bf load_a_frag(const float* rowp, int lane) {
  const int blo = (lane & 16) ? 8 : 0;
  const int bhi = (lane & 16) ? 24 : 16;
  v16bf a;
#pragma unroll
  for (int j = 0; j < 8; ++j) a[j] = (__bf16)rowp[blo + j];
#pragma unroll
  for (int j = 0; j < 8; ++j) a[8 + j] = (__bf16)rowp[bhi + j];
  return a;
}

// B fragment (32x16 bf16, K x N) from row-major weight W[N][ldw]:
// B[k][n] = W[n][k0 + k]. lane l holds col n0 + (l&15);
// lanes 0-15 hold K=0..7,16..23; lanes 16-31 hold K=8..15,24..31.
static __device__ __forceinline__ v16bf load_b_frag(const float* __restrict__ W,
                                                    int ldw, int n0, int k0,
                                                    int lane) {
  const float* p = W + (size_t)(n0 + (lane & 15)) * ldw + k0 + ((lane & 16) ? 8 : 0);
  v16bf b;
#pragma unroll
  for (int j = 0; j < 16; ++j) b[j] = (__bf16)p[(j < 8) ? j : (j + 8)];
  return b;
}

// ---------------------------------------------------------------------------
// Phase 1: gi = slow_frames @ W_ih^T + b_ih, stored TIME-MAJOR [NS][B][192]
// so the GRU streams one contiguous 12KB slab per (block, step).
// grid (7992, 3), block 128 (4 waves); wave = one 16x16 output tile.
// ---------------------------------------------------------------------------
__global__ __launch_bounds__(128) void k_gi(const float* __restrict__ x,
                                            const float* __restrict__ W_ih,
                                            const float* __restrict__ b_ih,
                                            float* __restrict__ gi) {
  const int lane  = threadIdx.x & 31;
  const int wave  = threadIdx.x >> 5;
  const int mtile = blockIdx.x;
  const int ntile = blockIdx.y * 4 + wave;            // 0..11
  const int n     = ntile * 16 + (lane & 15);

  const int m = lane & 15;
  const int r = mtile * 16 + m;                       // r = b*NS + s
  const int b = r / NS;
  const int s = r - b * NS;
  const float* xrow = x + (size_t)b * TLEN + (size_t)s * HOP_S;

  v8f acc = {};
#pragma unroll
  for (int kc = 0; kc < 3; ++kc) {
    v16bf a = load_a_frag(xrow + kc * 32, lane);
    v16bf w = load_b_frag(W_ih, LS, ntile * 16, kc * 32, lane);
    acc = wmma_bf16(a, w, acc);
  }
  const float bias = b_ih[n];
  const int hi = (lane & 16) ? 8 : 0;
#pragma unroll
  for (int v = 0; v < 8; ++v) {
    int rr = mtile * 16 + v + hi;                     // rr = bb*NS + ss
    int bb = rr / NS;
    int ss = rr - bb * NS;
    gi[((size_t)ss * BBATCH + bb) * NG + n] = acc[v] + bias;
  }
}

// ---------------------------------------------------------------------------
// Phase 2: GRU. 8 blocks (16 batch rows each), 4 waves per block.
// Wave w owns hidden cols 16w..16w+15 across gates r/z/n (tiles w, 4+w, 8+w)
// -> gate math elementwise in registers. Ping-pong LDS h => 1 barrier/step.
// gi loads for step t+1 issued during step t (hidden behind WMMA chain).
// hs stored time-major [NS][B][64].
// ---------------------------------------------------------------------------
__global__ __launch_bounds__(128) void k_gru(const float* __restrict__ gi,
                                             const float* __restrict__ W_hh,
                                             const float* __restrict__ b_hh,
                                             float* __restrict__ hs) {
  __shared__ float hbuf[2][16 * GHC];                  // ping-pong h [m][64]
  const int lane = threadIdx.x & 31;
  const int wave = threadIdx.x >> 5;                   // 0..3
  const int b0   = blockIdx.x * 16;
  const int nl   = lane & 15;
  const int col  = wave * 16 + nl;                     // owned hidden col
  const int hi   = (lane & 16) ? 8 : 0;
  const int m    = lane & 15;

  for (int i = threadIdx.x; i < 16 * GHC; i += blockDim.x) hbuf[0][i] = 0.0f;

  // Pin W_hh B-fragments (bf16) in VGPRs for the whole recurrence.
  const v16bf br0 = load_b_frag(W_hh, GHC, wave * 16,           0,  lane);
  const v16bf br1 = load_b_frag(W_hh, GHC, wave * 16,           32, lane);
  const v16bf bz0 = load_b_frag(W_hh, GHC, GHC + wave * 16,     0,  lane);
  const v16bf bz1 = load_b_frag(W_hh, GHC, GHC + wave * 16,     32, lane);
  const v16bf bn0 = load_b_frag(W_hh, GHC, 2 * GHC + wave * 16, 0,  lane);
  const v16bf bn1 = load_b_frag(W_hh, GHC, 2 * GHC + wave * 16, 32, lane);
  const float bhr = b_hh[wave * 16 + nl];
  const float bhz = b_hh[GHC + wave * 16 + nl];
  const float bhn = b_hh[2 * GHC + wave * 16 + nl];

  // Software pipeline: preload gi for t=0.
  float gir[8], giz[8], gin[8];
#pragma unroll
  for (int v = 0; v < 8; ++v) {
    size_t row = (size_t)(b0 + v + hi) * NG;           // t = 0
    gir[v] = gi[row + wave * 16 + nl];
    giz[v] = gi[row + GHC + wave * 16 + nl];
    gin[v] = gi[row + 2 * GHC + wave * 16 + nl];
  }
  __syncthreads();

  for (int t = 0; t < NS; ++t) {
    const int cur = t & 1, nxt = cur ^ 1;
    const float* hb = hbuf[cur];

    // Read h (A-layout fragments + the values this wave will update).
    v16bf a0 = load_a_frag(&hb[m * GHC + 0],  lane);
    v16bf a1 = load_a_frag(&hb[m * GHC + 32], lane);
    float hold[8];
#pragma unroll
    for (int v = 0; v < 8; ++v) hold[v] = hb[(v + hi) * GHC + col];

    // Issue next-step gi loads now; they drain behind the WMMA chain.
    const int tn = (t + 1 < NS) ? (t + 1) : t;
    float pgr[8], pgz[8], pgn[8];
#pragma unroll
    for (int v = 0; v < 8; ++v) {
      size_t row = ((size_t)tn * BBATCH + (b0 + v + hi)) * NG;
      pgr[v] = gi[row + wave * 16 + nl];
      pgz[v] = gi[row + GHC + wave * 16 + nl];
      pgn[v] = gi[row + 2 * GHC + wave * 16 + nl];
    }

    v8f gr = {}, gz = {}, gn = {};
    gr = wmma_bf16(a0, br0, gr); gr = wmma_bf16(a1, br1, gr);
    gz = wmma_bf16(a0, bz0, gz); gz = wmma_bf16(a1, bz1, gz);
    gn = wmma_bf16(a0, bn0, gn); gn = wmma_bf16(a1, bn1, gn);

#pragma unroll
    for (int v = 0; v < 8; ++v) {
      float rg = sigmoidf_(gir[v] + gr[v] + bhr);
      float zg = sigmoidf_(giz[v] + gz[v] + bhz);
      float ng = tanhf(gin[v] + rg * (gn[v] + bhn));
      float hn = (1.0f - zg) * ng + zg * hold[v];
      hbuf[nxt][(v + hi) * GHC + col] = hn;
      hs[((size_t)t * BBATCH + (b0 + v + hi)) * GHC + col] = hn;
    }
#pragma unroll
    for (int v = 0; v < 8; ++v) { gir[v] = pgr[v]; giz[v] = pgz[v]; gin[v] = pgn[v]; }
    __syncthreads();   // separates last reads of hbuf[cur] from its overwrite
  }
}

// ---------------------------------------------------------------------------
// Phase 3: cs = hs @ W_cs^T + b_cs, rows r' = t*B + b (time-major, per-row
// GEMM so row order is irrelevant). grid 7992, block 64 (2 waves, 2 N-tiles).
// ---------------------------------------------------------------------------
__global__ __launch_bounds__(64) void k_cs(const float* __restrict__ hs,
                                           const float* __restrict__ W_cs,
                                           const float* __restrict__ b_cs,
                                           float* __restrict__ cs) {
  const int lane  = threadIdx.x & 31;
  const int wave  = threadIdx.x >> 5;                  // 0..1
  const int mtile = blockIdx.x;
  const int n     = wave * 16 + (lane & 15);
  const int m     = lane & 15;
  const float* arow = hs + (size_t)(mtile * 16 + m) * GHC;

  v8f acc = {};
  acc = wmma_bf16(load_a_frag(arow, lane),
                  load_b_frag(W_cs, GHC, wave * 16, 0, lane), acc);
  acc = wmma_bf16(load_a_frag(arow + 32, lane),
                  load_b_frag(W_cs, GHC, wave * 16, 32, lane), acc);
  const float bias = b_cs[n];
  const int hi = (lane & 16) ? 8 : 0;
#pragma unroll
  for (int v = 0; v < 8; ++v)
    cs[(size_t)(mtile * 16 + v + hi) * CONDC + n] = acc[v] + bias;
}

// ---------------------------------------------------------------------------
// Phase 4: fused 2-layer MLP. One wave per 16-row M-tile; the 32+32 concat
// aligns with bf16 K-chunks. ReLU output staged via LDS to rebuild A-layout.
// grid 2999, block 256 (8 waves). cs is time-major: row = sidx*B + b.
// ---------------------------------------------------------------------------
__global__ __launch_bounds__(256) void k_mlp(const float* __restrict__ x,
                                             const float* __restrict__ cs,
                                             const float* __restrict__ W1,
                                             const float* __restrict__ b1,
                                             const float* __restrict__ W2,
                                             const float* __restrict__ b2,
                                             float* __restrict__ y) {
  __shared__ float hstg[8][16][GHC];                   // 32 KB
  const int lane  = threadIdx.x & 31;
  const int wave  = threadIdx.x >> 5;                  // 0..7
  const int mtile = blockIdx.x * 8 + wave;
  const int m     = lane & 15;
  const int hi    = (lane & 16) ? 8 : 0;

  const int r = mtile * 16 + m;                        // r = b*NF + f
  const int b = r / NF;
  const int f = r - b * NF;
  int sidx = f / DELTAC - 1; if (sidx < 0) sidx = 0;
  const float* fastrow = x + (size_t)b * TLEN + (size_t)f * HOPC;
  const float* condrow = cs + ((size_t)sidx * BBATCH + b) * CONDC;
  v16bf a0 = load_a_frag(fastrow, lane);               // K 0..31  = fast frame
  v16bf a1 = load_a_frag(condrow, lane);               // K 32..63 = cond

#pragma unroll
  for (int nt = 0; nt < 4; ++nt) {
    v8f acc = {};
    acc = wmma_bf16(a0, load_b_frag(W1, GHC, nt * 16, 0,  lane), acc);
    acc = wmma_bf16(a1, load_b_frag(W1, GHC, nt * 16, 32, lane), acc);
    const int n = nt * 16 + (lane & 15);
    const float bias = b1[n];
#pragma unroll
    for (int v = 0; v < 8; ++v) {
      float hv = acc[v] + bias;
      hstg[wave][v + hi][n] = hv > 0.0f ? hv : 0.0f;
    }
  }
  __syncthreads();                                     // order LDS D->A restage
  v16bf h0 = load_a_frag(&hstg[wave][m][0],  lane);
  v16bf h1 = load_a_frag(&hstg[wave][m][32], lane);

#pragma unroll
  for (int nt = 0; nt < 2; ++nt) {
    v8f acc = {};
    acc = wmma_bf16(h0, load_b_frag(W2, GHC, nt * 16, 0,  lane), acc);
    acc = wmma_bf16(h1, load_b_frag(W2, GHC, nt * 16, 32, lane), acc);
    const int n = nt * 16 + (lane & 15);
    const float bias = b2[n];
#pragma unroll
    for (int v = 0; v < 8; ++v)
      y[(size_t)(mtile * 16 + v + hi) * LF + n] = acc[v] + bias;
  }
}

// ---------------------------------------------------------------------------
// Phase 5: overlap-add as a gather (each sample covered by frames f2, f2-1).
// ---------------------------------------------------------------------------
__global__ __launch_bounds__(256) void k_ola(const float* __restrict__ y,
                                             float* __restrict__ out) {
  int i = blockIdx.x * blockDim.x + threadIdx.x;
  if (i >= BBATCH * TLEN) return;
  int b = i / TLEN;
  int t = i - b * TLEN;
  int f2 = t >> 4;
  float acc = 0.0f;
  if (f2 < NF) acc += y[((size_t)b * NF + f2) * LF + (t - (f2 << 4))];
  int f1 = f2 - 1;
  if (f1 >= 0) acc += y[((size_t)b * NF + f1) * LF + (t - (f1 << 4))];
  out[i] = acc;
}

// ---------------------------------------------------------------------------
extern "C" void kernel_launch(void* const* d_in, const int* in_sizes, int n_in,
                              void* d_out, int out_size, void* d_ws, size_t ws_size,
                              hipStream_t stream) {
  const float* x    = (const float*)d_in[0];
  const float* W_ih = (const float*)d_in[1];
  const float* W_hh = (const float*)d_in[2];
  const float* b_ih = (const float*)d_in[3];
  const float* b_hh = (const float*)d_in[4];
  const float* W_cs = (const float*)d_in[5];
  const float* b_cs = (const float*)d_in[6];
  const float* W1   = (const float*)d_in[7];
  const float* b1   = (const float*)d_in[8];
  const float* W2   = (const float*)d_in[9];
  const float* b2   = (const float*)d_in[10];
  float* out = (float*)d_out;

  // Workspace layout (gi is dead after k_gru; y aliases it):
  //   [0, 98.2MB)         gi  ([NS][B][192] f32)  /  y ([B*NF][32] f32, 49.1MB)
  //   [98.2MB, 131.0MB)   hs  ([NS][B][64] f32)
  //   [131.0MB, 147.3MB)  cs  ([NS][B][32] f32)
  const size_t gi_bytes = (size_t)BBATCH * NS * NG * sizeof(float);
  const size_t hs_bytes = (size_t)BBATCH * NS * GHC * sizeof(float);
  float* gi = (float*)d_ws;
  float* yb = (float*)d_ws;
  float* hs = (float*)((char*)d_ws + gi_bytes);
  float* cs = (float*)((char*)d_ws + gi_bytes + hs_bytes);

  const int mtiles_slow = (BBATCH * NS) / 16;          // 7992
  k_gi<<<dim3(mtiles_slow, 3), 128, 0, stream>>>(x, W_ih, b_ih, gi);
  k_gru<<<BBATCH / 16, 128, 0, stream>>>(gi, W_hh, b_hh, hs);
  k_cs<<<mtiles_slow, 64, 0, stream>>>(hs, W_cs, b_cs, cs);
  k_mlp<<<(BBATCH * NF) / (16 * 8), 256, 0, stream>>>(x, cs, W1, b1, W2, b2, yb);
  k_ola<<<(BBATCH * TLEN + 255) / 256, 256, 0, stream>>>(yb, out);
}
